// CREStereo_83399674954261
// MI455X (gfx1250) — compile-verified, matrix operand order checked by reading, not run on hardware
//
#include <hip/hip_runtime.h>
#include <math.h>

typedef _Float16 h16;
typedef _Float16 v16h __attribute__((ext_vector_type(16)));
typedef float    v8f  __attribute__((ext_vector_type(8)));

#define DI __device__ __forceinline__

DI float actf(float y, int act){
  if (act==1) return fmaxf(y,0.f);
  if (act==2) return 1.f/(1.f+expf(-y));
  if (act==3) return tanhf(y);
  return y;
}

DI float tapf(const float* p,int H,int W,int y,int x){
  if (x<0||x>W-1||y<0||y>H-1) return 0.f;
  return p[(size_t)y*W+x];
}

// kl (0..31 position within a K=32 chunk) -> fragment-major slot lh*16+e
DI int frag_off(int kl){
  return ((kl&7)|((kl>>4)<<3)) + (((kl>>3)&1)<<4);
}

// LDS row stride in halves: 48 (=96B, keeps 32B alignment for v16h reads)
#define LROW 48

// ------------------------------------------------------------------
// Weight prep: OIHW f32  ->  [Kp][Np] f16 panel, K = ((kh*KW+kw)*I+ci)
// ------------------------------------------------------------------
__global__ void k_prep_w(const float* __restrict__ w, h16* __restrict__ out,
                         int O,int I,int KH,int KW,int Kp,int Np){
  size_t idx=(size_t)blockIdx.x*blockDim.x+threadIdx.x;
  size_t total=(size_t)Kp*Np;
  if (idx>=total) return;
  int n=(int)(idx%Np); int k=(int)(idx/Np);
  int K=I*KH*KW;
  float v=0.f;
  if (n<O && k<K){
    int kh=k/(KW*I); int r=k-kh*KW*I; int kw=r/I; int ci=r-kw*I;
    v=w[(((size_t)n*I+ci)*KH+kh)*KW+kw];
  }
  out[idx]=(h16)v;
}

// ------------------------------------------------------------------
// Fragment helpers (LDS is fragment-major: contiguous 32B per fragment)
// ------------------------------------------------------------------
DI void frag_load(const h16* As,const h16* Bs,int wm,int wn,int lr,int lh,
                  v16h& af0,v16h& af1,v16h& bf0,v16h& bf1){
  af0=*(const v16h*)&As[(wm   +lr)*LROW+lh*16];
  af1=*(const v16h*)&As[(wm+16+lr)*LROW+lh*16];
  bf0=*(const v16h*)&Bs[(wn   +lr)*LROW+lh*16];
  bf1=*(const v16h*)&Bs[(wn+16+lr)*LROW+lh*16];
}
DI void frag_mma(const v16h& af0,const v16h& af1,const v16h& bf0,const v16h& bf1,
                 v8f acc[2][2]){
  acc[0][0]=__builtin_amdgcn_wmma_f32_16x16x32_f16(false,af0,false,bf0,(short)0,acc[0][0],false,false);
  acc[0][1]=__builtin_amdgcn_wmma_f32_16x16x32_f16(false,af0,false,bf1,(short)0,acc[0][1],false,false);
  acc[1][0]=__builtin_amdgcn_wmma_f32_16x16x32_f16(false,af1,false,bf0,(short)0,acc[1][0],false,false);
  acc[1][1]=__builtin_amdgcn_wmma_f32_16x16x32_f16(false,af1,false,bf1,(short)0,acc[1][1],false,false);
}

// ------------------------------------------------------------------
// Implicit-GEMM convolution (general), WMMA f32 = f16 x f16
// Double-buffered LDS, one barrier per K-tile.
// ------------------------------------------------------------------
__global__ __launch_bounds__(128)
void k_conv(const float* __restrict__ in, const h16* __restrict__ Bw,
            const float* __restrict__ bias, float* __restrict__ out,
            int Bn,int Cin,int H,int W,int Cout,int KH,int KW,
            int stride,int padH,int padW,int Ho,int Wo,int Kp,int Np,
            int act,float sc,float ad){
  __shared__ __align__(32) h16 As[2][64*LROW];
  __shared__ __align__(32) h16 Bs[2][64*LROW];
  const int HWin=H*W;
  const int M=Bn*Ho*Wo;
  const int blockM=blockIdx.x*64, blockN=blockIdx.y*64;
  const int tid=threadIdx.x, lane=tid&31, wv=tid>>5;
  const int wm=(wv>>1)*32, wn=(wv&1)*32;
  const int lr=lane&15, lh=lane>>4;
  const int KWC=KW*Cin, K=KH*KWC;

  // A-side: fixed K column per thread, rows m0+4e (decoded once)
  const int klA=tid&31;
  const int aST=(tid>>5)*LROW + frag_off(klA);
  int ohs[16], ows[16], inb[16];
  unsigned mmask=0u;
  {
    int m0=blockM+(tid>>5);
    #pragma unroll
    for(int e=0;e<16;e++){
      int m=m0+4*e;
      if(m<M){
        int ow=m%Wo; int tt=m/Wo; int oh=tt%Ho; int bb=tt/Ho;
        ohs[e]=oh*stride-padH;
        ows[e]=ow*stride-padW;
        inb[e]=bb*Cin*HWin;
        mmask|=1u<<e;
      } else { ohs[e]=0; ows[e]=0; inb[e]=0; }
    }
  }
  // B-side: fixed N column, kl = 2e+(tid>>6) (LDS slots loop-invariant)
  const int bc=tid&63;
  int stB[16];
  #pragma unroll
  for(int e=0;e<16;e++) stB[e]=bc*LROW + frag_off(e*2+(tid>>6));
  const size_t gB0=(size_t)(tid>>6)*Np + blockN + bc;

  auto fill=[&](int buf,int k0){
    int k=k0+klA;
    bool kok=(k<K);
    int kh=0,kw=0,ci=0;
    if(kok){ kh=k/KWC; int rem=k-kh*KWC; kw=rem/Cin; ci=rem-kw*Cin; }
    #pragma unroll
    for(int e=0;e<16;e++){
      float v=0.f;
      if(kok && (mmask&(1u<<e))){
        int ih=ohs[e]+kh, iw=ows[e]+kw;
        if((unsigned)ih<(unsigned)H && (unsigned)iw<(unsigned)W)
          v=in[(size_t)inb[e]+(size_t)ci*HWin+(size_t)ih*W+iw];
      }
      As[buf][aST+e*4*LROW]=(h16)v;
      Bs[buf][stB[e]]=Bw[(size_t)k0*Np+gB0+(size_t)e*2*Np];
    }
    if(k0+32<Kp) __builtin_prefetch(&Bw[(size_t)(k0+32)*Np+gB0],0,1);
  };

  v8f acc[2][2]={};
  const int nt=Kp>>5;
  fill(0,0);
  for(int i=0;i<nt;i++){
    __syncthreads();
    v16h af0,af1,bf0,bf1;
    frag_load(As[i&1],Bs[i&1],wm,wn,lr,lh,af0,af1,bf0,bf1);
    if(i+1<nt) fill((i+1)&1,(i+1)*32);
    frag_mma(af0,af1,bf0,bf1,acc);
  }
  #pragma unroll
  for(int mi=0;mi<2;mi++)
  #pragma unroll
  for(int ni=0;ni<2;ni++){
    int n=blockN+wn+ni*16+lr;
    if(n>=Cout) continue;
    #pragma unroll
    for(int vv=0;vv<8;vv++){
      int m=blockM+wm+mi*16+vv+8*lh;
      if(m>=M) continue;
      float y=acc[mi][ni][vv]+(bias?bias[n]:0.f);
      y=actf(y,act)*sc+ad;
      int ow=m%Wo; int tt=m/Wo; int oh=tt%Ho; int bb=tt/Ho;
      out[(((size_t)bb*Cout+n)*Ho+oh)*Wo+ow]=y;
    }
  }
}

// ------------------------------------------------------------------
// 1x1 / stride1 / pad0 convolution: pure pointer-arithmetic gather
// ------------------------------------------------------------------
__global__ __launch_bounds__(128)
void k_conv1x1(const float* __restrict__ in, const h16* __restrict__ Bw,
               const float* __restrict__ bias, float* __restrict__ out,
               int Bn,int Cin,int HW,int Cout,int Kp,int Np,
               int act,float sc,float ad){
  __shared__ __align__(32) h16 As[2][64*LROW];
  __shared__ __align__(32) h16 Bs[2][64*LROW];
  const int M=Bn*HW;
  const int blockM=blockIdx.x*64, blockN=blockIdx.y*64;
  const int tid=threadIdx.x, lane=tid&31, wv=tid>>5;
  const int wm=(wv>>1)*32, wn=(wv&1)*32;
  const int lr=lane&15, lh=lane>>4;

  const int klA=tid&31;
  const int aST=(tid>>5)*LROW + frag_off(klA);
  size_t inoff[16];
  unsigned mmask=0u;
  {
    int m0=blockM+(tid>>5);
    #pragma unroll
    for(int e=0;e<16;e++){
      int m=m0+4*e;
      if(m<M){
        int sp=m%HW; int bb=m/HW;
        inoff[e]=(size_t)bb*Cin*HW+sp;
        mmask|=1u<<e;
      } else inoff[e]=0;
    }
  }
  const int bc=tid&63;
  int stB[16];
  #pragma unroll
  for(int e=0;e<16;e++) stB[e]=bc*LROW + frag_off(e*2+(tid>>6));
  const size_t gB0=(size_t)(tid>>6)*Np + blockN + bc;

  auto fill=[&](int buf,int k0){
    int k=k0+klA;
    bool kok=(k<Cin);
    #pragma unroll
    for(int e=0;e<16;e++){
      float v=0.f;
      if(kok && (mmask&(1u<<e))) v=in[inoff[e]+(size_t)k*HW];
      As[buf][aST+e*4*LROW]=(h16)v;
      Bs[buf][stB[e]]=Bw[(size_t)k0*Np+gB0+(size_t)e*2*Np];
    }
    if(k0+32<Kp) __builtin_prefetch(&Bw[(size_t)(k0+32)*Np+gB0],0,1);
  };

  v8f acc[2][2]={};
  const int nt=Kp>>5;
  fill(0,0);
  for(int i=0;i<nt;i++){
    __syncthreads();
    v16h af0,af1,bf0,bf1;
    frag_load(As[i&1],Bs[i&1],wm,wn,lr,lh,af0,af1,bf0,bf1);
    if(i+1<nt) fill((i+1)&1,(i+1)*32);
    frag_mma(af0,af1,bf0,bf1,acc);
  }
  #pragma unroll
  for(int mi=0;mi<2;mi++)
  #pragma unroll
  for(int ni=0;ni<2;ni++){
    int n=blockN+wn+ni*16+lr;
    if(n>=Cout) continue;
    #pragma unroll
    for(int vv=0;vv<8;vv++){
      int m=blockM+wm+mi*16+vv+8*lh;
      if(m>=M) continue;
      float y=acc[mi][ni][vv]+(bias?bias[n]:0.f);
      y=actf(y,act)*sc+ad;
      int sp=m%HW; int bb=m/HW;
      out[(size_t)(bb*Cout+n)*HW+sp]=y;
    }
  }
}

// ------------------------------------------------------------------
// Dense GEMM (tokens x weight-panel), WMMA f16, C row-major f32
// ------------------------------------------------------------------
__global__ __launch_bounds__(128)
void k_gemm(const float* __restrict__ A, const h16* __restrict__ Bw,
            float* __restrict__ C, int M,int K,int N,int Kp,int Np,int act){
  __shared__ __align__(32) h16 As[2][64*LROW];
  __shared__ __align__(32) h16 Bs[2][64*LROW];
  const int blockM=blockIdx.x*64, blockN=blockIdx.y*64;
  const int tid=threadIdx.x, lane=tid&31, wv=tid>>5;
  const int wm=(wv>>1)*32, wn=(wv&1)*32;
  const int lr=lane&15, lh=lane>>4;

  const int klA=tid&31;
  const int aST=(tid>>5)*LROW + frag_off(klA);
  int am[16]; unsigned mmask=0u;
  {
    int m0=blockM+(tid>>5);
    #pragma unroll
    for(int e=0;e<16;e++){
      int m=m0+4*e;
      am[e]=m;
      if(m<M) mmask|=1u<<e;
    }
  }
  const int bc=tid&63;
  int stB[16];
  #pragma unroll
  for(int e=0;e<16;e++) stB[e]=bc*LROW + frag_off(e*2+(tid>>6));
  const size_t gB0=(size_t)(tid>>6)*Np + blockN + bc;

  auto fill=[&](int buf,int k0){
    int k=k0+klA;
    bool kok=(k<K);
    #pragma unroll
    for(int e=0;e<16;e++){
      float v=0.f;
      if(kok && (mmask&(1u<<e))) v=A[(size_t)am[e]*K+k];
      As[buf][aST+e*4*LROW]=(h16)v;
      Bs[buf][stB[e]]=Bw[(size_t)k0*Np+gB0+(size_t)e*2*Np];
    }
    if(k0+32<Kp) __builtin_prefetch(&Bw[(size_t)(k0+32)*Np+gB0],0,1);
  };

  v8f acc[2][2]={};
  const int nt=Kp>>5;
  fill(0,0);
  for(int i=0;i<nt;i++){
    __syncthreads();
    v16h af0,af1,bf0,bf1;
    frag_load(As[i&1],Bs[i&1],wm,wn,lr,lh,af0,af1,bf0,bf1);
    if(i+1<nt) fill((i+1)&1,(i+1)*32);
    frag_mma(af0,af1,bf0,bf1,acc);
  }
  #pragma unroll
  for(int mi=0;mi<2;mi++)
  #pragma unroll
  for(int ni=0;ni<2;ni++){
    int n=blockN+wn+ni*16+lr;
    if(n>=N) continue;
    #pragma unroll
    for(int vv=0;vv<8;vv++){
      int m=blockM+wm+mi*16+vv+8*lh;
      if(m>=M) continue;
      C[(size_t)m*N+n]=actf(acc[mi][ni][vv],act);
    }
  }
}

// ------------------------------------------------------------------
// Normalizations / elementwise
// ------------------------------------------------------------------
__global__ __launch_bounds__(256)
void k_instnorm(float* x, int HW, int relu){
  float* p=x+(size_t)blockIdx.x*HW;
  __shared__ float red[256];
  int t=threadIdx.x;
  float s=0.f;
  for(int i=t;i<HW;i+=256) s+=p[i];
  red[t]=s; __syncthreads();
  for(int o=128;o;o>>=1){ if(t<o) red[t]+=red[t+o]; __syncthreads(); }
  float mean=red[0]/HW;
  __syncthreads();
  float v=0.f;
  for(int i=t;i<HW;i+=256){ float d=p[i]-mean; v+=d*d; }
  red[t]=v; __syncthreads();
  for(int o=128;o;o>>=1){ if(t<o) red[t]+=red[t+o]; __syncthreads(); }
  float inv=rsqrtf(red[0]/HW+1e-5f);
  for(int i=t;i<HW;i+=256){
    float y=(p[i]-mean)*inv;
    if(relu) y=fmaxf(y,0.f);
    p[i]=y;
  }
}

__global__ __launch_bounds__(256)
void k_layernorm(float* x, const float* w, const float* b){
  float* row=x+(size_t)blockIdx.x*256;
  __shared__ float red[256];
  int t=threadIdx.x;
  float v=row[t];
  red[t]=v; __syncthreads();
  for(int o=128;o;o>>=1){ if(t<o) red[t]+=red[t+o]; __syncthreads(); }
  float mean=red[0]*(1.f/256.f);
  __syncthreads();
  float d=v-mean; red[t]=d*d; __syncthreads();
  for(int o=128;o;o>>=1){ if(t<o) red[t]+=red[t+o]; __syncthreads(); }
  float inv=rsqrtf(red[0]*(1.f/256.f)+1e-5f);
  row[t]=d*inv*w[t]+b[t];
}

__global__ void k_imnorm(const float* a,const float* b,float* out,size_t per){
  size_t i=(size_t)blockIdx.x*blockDim.x+threadIdx.x;
  if(i>=2*per) return;
  float v=(i<per)?a[i]:b[i-per];
  out[i]=2.f*(v/255.f)-1.f;
}

__global__ void k_avgpool(const float* in,float* out,int NC,int H,int W,int k){
  int Ho=H/k, Wo=W/k;
  size_t total=(size_t)NC*Ho*Wo;
  size_t i=(size_t)blockIdx.x*blockDim.x+threadIdx.x;
  if(i>=total) return;
  int wo=(int)(i%Wo); size_t t=i/Wo; int ho=(int)(t%Ho); int p=(int)(t/Ho);
  const float* src=in+(size_t)p*H*W;
  float s=0.f;
  for(int a=0;a<k;a++) for(int b=0;b<k;b++) s+=src[(size_t)(ho*k+a)*W+wo*k+b];
  out[i]=s/(k*k);
}

__global__ void k_slice_act(const float* in,float* out,int N,int Ctot,int c0,int C,int HW,int mode){
  size_t total=(size_t)N*C*HW;
  size_t idx=(size_t)blockIdx.x*blockDim.x+threadIdx.x;
  if(idx>=total) return;
  int i=(int)(idx%HW); size_t t=idx/HW; int c=(int)(t%C); int n=(int)(t/C);
  out[idx]=actf(in[((size_t)(n*Ctot+c0+c))*HW+i],mode);
}

__global__ void k_concat2(const float* a,int Ca,const float* b,int Cb,float* out,int N,int HW){
  int C=Ca+Cb;
  size_t total=(size_t)N*C*HW;
  size_t idx=(size_t)blockIdx.x*blockDim.x+threadIdx.x;
  if(idx>=total) return;
  int i=(int)(idx%HW); size_t t=idx/HW; int c=(int)(t%C); int n=(int)(t/C);
  out[idx]=(c<Ca)? a[((size_t)(n*Ca+c))*HW+i] : b[((size_t)(n*Cb+(c-Ca)))*HW+i];
}

__global__ void k_concat_last(const float* a,int Da,const float* b,int Db,float* out,int rows){
  int D=Da+Db;
  size_t total=(size_t)rows*D;
  size_t idx=(size_t)blockIdx.x*blockDim.x+threadIdx.x;
  if(idx>=total) return;
  int d=(int)(idx%D); int r=(int)(idx/D);
  out[idx]=(d<Da)? a[(size_t)r*Da+d] : b[(size_t)r*Db+(d-Da)];
}

__global__ void k_add2(float* out,const float* a,const float* b,size_t n){
  size_t i=(size_t)blockIdx.x*blockDim.x+threadIdx.x;
  if(i<n) out[i]=a[i]+b[i];
}
__global__ void k_add_relu(float* out,const float* a,const float* b,size_t n){
  size_t i=(size_t)blockIdx.x*blockDim.x+threadIdx.x;
  if(i<n) out[i]=fmaxf(a[i]+b[i],0.f);
}
__global__ void k_addto(float* x,const float* y,size_t n){
  size_t i=(size_t)blockIdx.x*blockDim.x+threadIdx.x;
  if(i<n) x[i]+=y[i];
}
__global__ void k_mul(float* out,const float* a,const float* b,size_t n){
  size_t i=(size_t)blockIdx.x*blockDim.x+threadIdx.x;
  if(i<n) out[i]=a[i]*b[i];
}
__global__ void k_gru_combine(float* h,const float* z,const float* q,size_t n){
  size_t i=(size_t)blockIdx.x*blockDim.x+threadIdx.x;
  if(i<n){ float zz=z[i]; h[i]=(1.f-zz)*h[i]+zz*q[i]; }
}
__global__ void k_zero(float* p,size_t n){
  size_t i=(size_t)blockIdx.x*blockDim.x+threadIdx.x;
  if(i<n) p[i]=0.f;
}
__global__ void k_elu1(const float* in,float* out,size_t n){
  size_t i=(size_t)blockIdx.x*blockDim.x+threadIdx.x;
  if(i<n){ float x=in[i]; out[i]=(x>0.f)? x+1.f : expf(x); }
}

// ------------------------------------------------------------------
// Linear attention pieces (8 heads x 32 dims, D=256)
// ------------------------------------------------------------------
__global__ void k_ksum(const float* k_,float* ks,int N,int L){
  int idx=blockIdx.x*blockDim.x+threadIdx.x;
  if(idx>=N*256) return;
  int d=idx&31; int hh=(idx>>5)&7; int n=idx>>8;
  float s=0.f;
  for(int ss=0;ss<L;ss++) s+=k_[((size_t)(n*L+ss))*256+hh*32+d];
  ks[idx]=s;
}
__global__ void k_kv(const float* ke,const float* vb,float* kv,int N,int L){
  int idx=blockIdx.x*blockDim.x+threadIdx.x;
  if(idx>=N*8192) return;
  int v=idx&31; int d=(idx>>5)&31; int hh=(idx>>10)&7; int n=idx>>13;
  float s=0.f;
  for(int ss=0;ss<L;ss++){
    size_t base=((size_t)(n*L+ss))*256+hh*32;
    s+=ke[base+d]*vb[base+v];
  }
  kv[idx]=s;
}
__global__ void k_attn_out(const float* qe,const float* kv,const float* ks,float* out,int N,int L){
  int idx=blockIdx.x*blockDim.x+threadIdx.x;
  if(idx>=N*L*8) return;
  int hh=idx&7; int l=(idx>>3)%L; int n=idx/(8*L);
  const float* qrow=qe+((size_t)(n*L+l))*256+hh*32;
  const float* ksr=ks+(size_t)(n*8+hh)*32;
  float den=1e-6f;
  for(int d=0;d<32;d++) den+=qrow[d]*ksr[d];
  float z=1.f/den;
  const float* kvp=kv+((size_t)(n*8+hh))*1024;
  float* orow=out+((size_t)(n*L+l))*256+hh*32;
  for(int v=0;v<32;v++){
    float s=0.f;
    for(int d=0;d<32;d++) s+=qrow[d]*kvp[d*32+v];
    orow[v]=s*z;
  }
}

// ------------------------------------------------------------------
// Position encoding -> tokens ; tokens -> NCHW
// ------------------------------------------------------------------
__global__ void k_posenc_tok(const float* f,float* tok,int N,int C,int H,int W){
  size_t total=(size_t)N*H*W*C;
  size_t idx=(size_t)blockIdx.x*blockDim.x+threadIdx.x;
  if(idx>=total) return;
  int c=(int)(idx%C); size_t t=idx/C; int w=(int)(t%W); t/=W; int h=(int)(t%H); int n=(int)(t/H);
  int i=c>>2; int g=c&3;
  float div=expf((2.f*i)*(-9.210340371976184f/128.f));
  float pos=(g<2)? (w+1.f) : (h+1.f);
  float pe=((g&1)==0)? sinf(pos*div) : cosf(pos*div);
  tok[idx]=f[(((size_t)n*C+c)*H+h)*W+w]+pe;
}
__global__ void k_tok2nchw(const float* tok,float* f,int N,int C,int H,int W){
  size_t total=(size_t)N*C*H*W;
  size_t idx=(size_t)blockIdx.x*blockDim.x+threadIdx.x;
  if(idx>=total) return;
  int w=(int)(idx%W); size_t t=idx/W; int h=(int)(t%H); t/=H; int c=(int)(t%C); int n=(int)(t/C);
  f[idx]=tok[(((size_t)(n*H+h))*W+w)*256+c];
}

// ------------------------------------------------------------------
// Correlation / sampling / upsampling
// ------------------------------------------------------------------
__global__ void k_corr_att(const float* f1,const float* f2,const float* flow,const float* eo,
                           float* corr,int N,int C,int H,int W,int small){
  size_t total=(size_t)N*36*H*W;
  size_t idx=(size_t)blockIdx.x*blockDim.x+threadIdx.x;
  if(idx>=total) return;
  int w=(int)(idx%W); size_t t=idx/W; int h=(int)(t%H); t/=H; int gk=(int)(t%36); int n=(int)(t/36);
  int g=gk/9, k=gk%9;
  int rx= small?1:4, ry= small?1:0;
  int pw=2*rx+1;
  int dx=k%pw-rx, dy=k/pw-ry;
  size_t hw=(size_t)H*W;
  float fx=flow[((size_t)n*2+0)*hw+(size_t)h*W+w];
  float fy=flow[((size_t)n*2+1)*hw+(size_t)h*W+w];
  float ex=eo[((size_t)n*18+k*2+0)*hw+(size_t)h*W+w];
  float ey=eo[((size_t)n*18+k*2+1)*hw+(size_t)h*W+w];
  float cx=w+fx+ex+dx, cy=h+fy+ey+dy;
  float x0f=floorf(cx), y0f=floorf(cy);
  int x0=(int)x0f, y0=(int)y0f;
  float wx=cx-x0f, wy=cy-y0f;
  int Cg=C/4;
  float s=0.f;
  for(int c=0;c<Cg;c++){
    int cc=g*Cg+c;
    const float* img=f2+((size_t)n*C+cc)*hw;
    float v =tapf(img,H,W,y0  ,x0  )*(1.f-wx)*(1.f-wy);
    v+=tapf(img,H,W,y0  ,x0+1)*wx*(1.f-wy);
    v+=tapf(img,H,W,y0+1,x0  )*(1.f-wx)*wy;
    v+=tapf(img,H,W,y0+1,x0+1)*wx*wy;
    s+=v*f1[((size_t)n*C+cc)*hw+(size_t)h*W+w];
  }
  corr[idx]=s/Cg;
}

__global__ void k_bilin(const float* img,const float* flow,float* out,int N,int C,int H,int W){
  size_t total=(size_t)N*C*H*W;
  size_t idx=(size_t)blockIdx.x*blockDim.x+threadIdx.x;
  if(idx>=total) return;
  int w=(int)(idx%W); size_t t=idx/W; int h=(int)(t%H); t/=H; int c=(int)(t%C); int n=(int)(t/C);
  size_t hw=(size_t)H*W;
  float cx=w+flow[((size_t)n*2+0)*hw+(size_t)h*W+w];
  float cy=h+flow[((size_t)n*2+1)*hw+(size_t)h*W+w];
  float x0f=floorf(cx), y0f=floorf(cy);
  int x0=(int)x0f, y0=(int)y0f;
  float wx=cx-x0f, wy=cy-y0f;
  const float* p=img+((size_t)n*C+c)*hw;
  float v =tapf(p,H,W,y0  ,x0  )*(1.f-wx)*(1.f-wy);
  v+=tapf(p,H,W,y0  ,x0+1)*wx*(1.f-wy);
  v+=tapf(p,H,W,y0+1,x0  )*(1.f-wx)*wy;
  v+=tapf(p,H,W,y0+1,x0+1)*wx*wy;
  out[idx]=v;
}

__global__ void k_corr_patch(const float* left,const float* right,float* corr,
                             int N,int C,int H,int W,int small){
  size_t total=(size_t)N*36*H*W;
  size_t idx=(size_t)blockIdx.x*blockDim.x+threadIdx.x;
  if(idx>=total) return;
  int w=(int)(idx%W); size_t t=idx/W; int h=(int)(t%H); t/=H; int gk=(int)(t%36); int n=(int)(t/36);
  int g=gk/9, k=gk%9;
  int PW= small?3:9;
  int pady= small?1:0, padx=PW/2;
  int ph=k/PW, pw=k%PW;
  int hh=min(max(h+ph-pady,0),H-1);
  int ww=min(max(w+pw-padx,0),W-1);
  int Cg=C/4;
  size_t hw=(size_t)H*W;
  float s=0.f;
  for(int c=0;c<Cg;c++){
    int cc=g*Cg+c;
    s+=left[((size_t)n*C+cc)*hw+(size_t)h*W+w]*right[((size_t)n*C+cc)*hw+(size_t)hh*W+ww];
  }
  corr[idx]=s/Cg;
}

__global__ void k_upsample(const float* flow,const float* mask,float* out,
                           int N,int H,int W,float sgn){
  int H4=H*4, W4=W*4;
  size_t total=(size_t)N*2*H4*W4;
  size_t idx=(size_t)blockIdx.x*blockDim.x+threadIdx.x;
  if(idx>=total) return;
  int w4=(int)(idx%W4); size_t t=idx/W4; int h4=(int)(t%H4); t/=H4; int d=(int)(t%2); int n=(int)(t/2);
  int h=h4>>2, r1=h4&3, w=w4>>2, r2=w4&3;
  float mv[9]; float mx=-1e30f;
  for(int k=0;k<9;k++){
    mv[k]=mask[((size_t)(n*144+k*16+r1*4+r2)*H+h)*W+w];
    mx=fmaxf(mx,mv[k]);
  }
  float se=0.f;
  for(int k=0;k<9;k++){ mv[k]=expf(mv[k]-mx); se+=mv[k]; }
  float acc=0.f;
  for(int k=0;k<9;k++){
    int dy=k/3-1, dx=k%3-1;
    int hh=h+dy, ww=w+dx;
    float fv=(hh>=0&&hh<H&&ww>=0&&ww<W)? flow[((size_t)(n*2+d)*H+hh)*W+ww] : 0.f;
    acc+=(mv[k]/se)*4.f*fv;
  }
  out[idx]=sgn*acc;
}

__global__ void k_interp(const float* in,float* out,int NC,int Hin,int Win,int Hout,int Wout,float scale){
  size_t total=(size_t)NC*Hout*Wout;
  size_t idx=(size_t)blockIdx.x*blockDim.x+threadIdx.x;
  if(idx>=total) return;
  int wo=(int)(idx%Wout); size_t t=idx/Wout; int ho=(int)(t%Hout); int p=(int)(t/Hout);
  float py=(Hout>1)? ho*(float)(Hin-1)/(Hout-1):0.f;
  float px=(Wout>1)? wo*(float)(Win-1)/(Wout-1):0.f;
  float fy=floorf(py); if(fy>Hin-2) fy=(float)(Hin-2); if(fy<0) fy=0.f;
  float fx=floorf(px); if(fx>Win-2) fx=(float)(Win-2); if(fx<0) fx=0.f;
  int y0=(int)fy, x0=(int)fx;
  float wy=py-fy, wx=px-fx;
  const float* q=in+(size_t)p*Hin*Win;
  float v=(1.f-wy)*((1.f-wx)*q[(size_t)y0*Win+x0]+wx*q[(size_t)y0*Win+x0+1])
         +wy*((1.f-wx)*q[(size_t)(y0+1)*Win+x0]+wx*q[(size_t)(y0+1)*Win+x0+1]);
  out[idx]=scale*v;
}

// ==================================================================
// Host side
// ==================================================================
static inline unsigned g1(size_t n){ return (unsigned)((n+255)/256); }

struct Arena{
  char* base; size_t off;
  float* f(size_t n){ size_t a=(off+255)&~(size_t)255; off=a+n*sizeof(float); return (float*)(base+a); }
  h16*  h(size_t n){ size_t a=(off+255)&~(size_t)255; off=a+n*sizeof(h16);  return (h16*)(base+a); }
};

struct ConvW{ const h16* w; const float* b; int O,I,KH,KW,Kp,Np; };

static ConvW mkconv(Arena& ar, hipStream_t st, const float* w, const float* b,
                    int O,int I,int KH,int KW){
  ConvW c; c.b=b; c.O=O; c.I=I; c.KH=KH; c.KW=KW;
  c.Kp=((I*KH*KW)+31)/32*32;
  c.Np=(O+63)/64*64;
  h16* d=ar.h((size_t)c.Kp*c.Np);
  size_t n=(size_t)c.Kp*c.Np;
  k_prep_w<<<g1(n),256,0,st>>>(w,d,O,I,KH,KW,c.Kp,c.Np);
  c.w=d;
  return c;
}

static void conv(hipStream_t st,const float* in,const ConvW& c,float* out,
                 int B,int H,int W,int stride,int padH,int padW,
                 int act=0,float sc=1.f,float ad=0.f){
  int Ho=(H+2*padH-c.KH)/stride+1, Wo=(W+2*padW-c.KW)/stride+1;
  int M=B*Ho*Wo;
  dim3 grid((M+63)/64, c.Np/64);
  if(c.KH==1 && c.KW==1 && stride==1 && padH==0 && padW==0)
    k_conv1x1<<<grid,128,0,st>>>(in,c.w,c.b,out,B,c.I,H*W,c.O,c.Kp,c.Np,act,sc,ad);
  else
    k_conv<<<grid,128,0,st>>>(in,c.w,c.b,out,B,c.I,H,W,c.O,c.KH,c.KW,
                              stride,padH,padW,Ho,Wo,c.Kp,c.Np,act,sc,ad);
}

static void gemm(hipStream_t st,const float* A,const ConvW& w,float* C,int M,int act){
  dim3 grid((M+63)/64, w.Np/64);
  k_gemm<<<grid,128,0,st>>>(A,w.w,C,M,w.I,w.O,w.Kp,w.Np,act);
}

static void inst(hipStream_t st,float* x,int planes,int hw,int relu){
  k_instnorm<<<planes,256,0,st>>>(x,hw,relu);
}

static void resblock(hipStream_t st,float* x,float* t1,float* t2,float* sc,
                     const ConvW& c1,const ConvW& c2,const ConvW* dw,
                     int Bn,int& H,int& W,int stride){
  int Ho=(H+2-3)/stride+1, Wo=(W+2-3)/stride+1;
  conv(st,x,c1,t1,Bn,H,W,stride,1,1,0);
  inst(st,t1,Bn*c1.O,Ho*Wo,1);
  conv(st,t1,c2,t2,Bn,Ho,Wo,1,1,1,0);
  inst(st,t2,Bn*c2.O,Ho*Wo,1);
  size_t n=(size_t)Bn*c2.O*Ho*Wo;
  if(dw){
    conv(st,x,*dw,sc,Bn,H,W,stride,0,0,0);
    inst(st,sc,Bn*dw->O,Ho*Wo,0);
    k_add_relu<<<g1(n),256,0,st>>>(x,sc,t2,n);
  } else {
    k_add_relu<<<g1(n),256,0,st>>>(x,x,t2,n);
  }
  H=Ho; W=Wo;
}

struct AttW{ ConvW q,k,v,m,m1,m2; const float *ln1w,*ln1b,*ln2w,*ln2b; };
struct TokS{ float *q,*k,*v,*qe,*ke,*ks,*kv,*ao,*m,*cat,*h1,*m2b; };

static void loftr(hipStream_t st,const AttW& A,TokS& S,
                  const float* x,const float* src,float* out,int N,int L){
  int rows=N*L;
  size_t nt=(size_t)rows*256;
  gemm(st,x  ,A.q,S.q,rows,0);
  gemm(st,src,A.k,S.k,rows,0);
  gemm(st,src,A.v,S.v,rows,0);
  k_elu1<<<g1(nt),256,0,st>>>(S.q,S.qe,nt);
  k_elu1<<<g1(nt),256,0,st>>>(S.k,S.ke,nt);
  k_ksum<<<g1((size_t)N*256),256,0,st>>>(S.ke,S.ks,N,L);
  k_kv<<<g1((size_t)N*8192),256,0,st>>>(S.ke,S.v,S.kv,N,L);
  k_attn_out<<<g1((size_t)N*L*8),256,0,st>>>(S.qe,S.kv,S.ks,S.ao,N,L);
  gemm(st,S.ao,A.m,S.m,rows,0);
  k_layernorm<<<rows,256,0,st>>>(S.m,A.ln1w,A.ln1b);
  k_concat_last<<<g1((size_t)rows*512),256,0,st>>>(x,256,S.m,256,S.cat,rows);
  gemm(st,S.cat,A.m1,S.h1,rows,1);
  gemm(st,S.h1 ,A.m2,S.m2b,rows,0);
  k_layernorm<<<rows,256,0,st>>>(S.m2b,A.ln2w,A.ln2b);
  k_add2<<<g1(nt),256,0,st>>>(out,x,S.m2b,nt);
}

struct UpdW{ ConvW c1,c2,f1,f2,cw,z1,r1,q1,z2,r2,q2,fh1,fh2,m1,m2; };
struct Tmp{ float *corr,*t256a,*t256b,*t192,*t128a,*t128b,*mf,*hx,*z,*r,*q,*mask,*dflow; };

static void gru_pass(hipStream_t st,const ConvW& zw,const ConvW& rw,const ConvW& qw,
                     float* h,const float* x,Tmp& T,int N,int H,int W,int padH,int padW){
  int HW=H*W;
  size_t n128=(size_t)N*128*HW, n384=(size_t)N*384*HW;
  k_concat2<<<g1(n384),256,0,st>>>(h,128,x,256,T.hx,N,HW);
  conv(st,T.hx,zw,T.z,N,H,W,1,padH,padW,2);
  conv(st,T.hx,rw,T.r,N,H,W,1,padH,padW,2);
  k_mul<<<g1(n128),256,0,st>>>(T.t128b,T.r,h,n128);
  k_concat2<<<g1(n384),256,0,st>>>(T.t128b,128,x,256,T.hx,N,HW);
  conv(st,T.hx,qw,T.q,N,H,W,1,padH,padW,3);
  k_gru_combine<<<g1(n128),256,0,st>>>(h,T.z,T.q,n128);
}

static void update_block(hipStream_t st,const UpdW& U,Tmp& T,
                         float* net,const float* inp,const float* corr,const float* flow,
                         int N,int H,int W,bool doMask){
  int HW=H*W;
  conv(st,corr   ,U.c1,T.t256a,N,H,W,1,0,0,1);
  conv(st,T.t256a,U.c2,T.t192 ,N,H,W,1,1,1,1);
  conv(st,flow   ,U.f1,T.t128a,N,H,W,1,3,3,1);
  conv(st,T.t128a,U.f2,T.t128b,N,H,W,1,1,1,1);
  k_concat2<<<g1((size_t)N*256*HW),256,0,st>>>(T.t192,192,T.t128b,64,T.t256a,N,HW);
  conv(st,T.t256a,U.cw,T.t128a,N,H,W,1,1,1,1);                      // 126 ch
  k_concat2<<<g1((size_t)N*128*HW),256,0,st>>>(T.t128a,126,flow,2,T.mf,N,HW);
  k_concat2<<<g1((size_t)N*256*HW),256,0,st>>>(inp,128,T.mf,128,T.t256b,N,HW);
  gru_pass(st,U.z1,U.r1,U.q1,net,T.t256b,T,N,H,W,0,2);              // 1x5
  gru_pass(st,U.z2,U.r2,U.q2,net,T.t256b,T,N,H,W,2,0);              // 5x1
  conv(st,net,U.fh1,T.t256b,N,H,W,1,1,1,1);
  conv(st,T.t256b,U.fh2,T.dflow,N,H,W,1,1,1,0);
  if(doMask){
    conv(st,net,U.m1,T.t256a,N,H,W,1,1,1,1);
    conv(st,T.t256a,U.m2,T.mask,N,H,W,1,0,0,0,0.25f,0.f);
  }
}

extern "C" void kernel_launch(void* const* d_in, const int* in_sizes, int n_in,
                              void* d_out, int out_size, void* d_ws, size_t ws_size,
                              hipStream_t stream){
  (void)in_sizes; (void)n_in; (void)out_size; (void)ws_size;
  hipStream_t st=stream;
  Arena ar{(char*)d_ws,0};
#define P(i) ((const float*)d_in[(i)])
  const float* img1=P(0);
  const float* img2=P(1);

  // ---- fp16 weight panels (prepped every call; deterministic) ----
  ConvW c_conv1=mkconv(ar,st,P(2) ,P(3) ,64 ,3  ,7,7);
  ConvW l1b1c1 =mkconv(ar,st,P(4) ,P(5) ,64 ,64 ,3,3);
  ConvW l1b1c2 =mkconv(ar,st,P(6) ,P(7) ,64 ,64 ,3,3);
  ConvW l1b2c1 =mkconv(ar,st,P(8) ,P(9) ,64 ,64 ,3,3);
  ConvW l1b2c2 =mkconv(ar,st,P(10),P(11),64 ,64 ,3,3);
  ConvW l2b1c1 =mkconv(ar,st,P(12),P(13),96 ,64 ,3,3);
  ConvW l2b1c2 =mkconv(ar,st,P(14),P(15),96 ,96 ,3,3);
  ConvW l2b1d  =mkconv(ar,st,P(16),P(17),96 ,64 ,1,1);
  ConvW l2b2c1 =mkconv(ar,st,P(18),P(19),96 ,96 ,3,3);
  ConvW l2b2c2 =mkconv(ar,st,P(20),P(21),96 ,96 ,3,3);
  ConvW l3b1c1 =mkconv(ar,st,P(22),P(23),128,96 ,3,3);
  ConvW l3b1c2 =mkconv(ar,st,P(24),P(25),128,128,3,3);
  ConvW l3b1d  =mkconv(ar,st,P(26),P(27),128,96 ,1,1);
  ConvW l3b2c1 =mkconv(ar,st,P(28),P(29),128,128,3,3);
  ConvW l3b2c2 =mkconv(ar,st,P(30),P(31),128,128,3,3);
  ConvW c_conv2=mkconv(ar,st,P(32),P(33),256,128,1,1);
  UpdW U;
  U.c1 =mkconv(ar,st,P(34),P(35),256,36 ,1,1);
  U.c2 =mkconv(ar,st,P(36),P(37),192,256,3,3);
  U.f1 =mkconv(ar,st,P(38),P(39),128,2  ,7,7);
  U.f2 =mkconv(ar,st,P(40),P(41),64 ,128,3,3);
  U.cw =mkconv(ar,st,P(42),P(43),126,256,3,3);
  U.z1 =mkconv(ar,st,P(44),P(45),128,384,1,5);
  U.r1 =mkconv(ar,st,P(46),P(47),128,384,1,5);
  U.q1 =mkconv(ar,st,P(48),P(49),128,384,1,5);
  U.z2 =mkconv(ar,st,P(50),P(51),128,384,5,1);
  U.r2 =mkconv(ar,st,P(52),P(53),128,384,5,1);
  U.q2 =mkconv(ar,st,P(54),P(55),128,384,5,1);
  U.fh1=mkconv(ar,st,P(56),P(57),256,128,3,3);
  U.fh2=mkconv(ar,st,P(58),P(59),2  ,256,3,3);
  U.m1 =mkconv(ar,st,P(60),P(61),256,128,3,3);
  U.m2 =mkconv(ar,st,P(62),P(63),144,256,1,1);
  AttW SA, CA;
  {
    int i0=64;
    SA.q =mkconv(ar,st,P(i0+0),nullptr,256,256,1,1);
    SA.k =mkconv(ar,st,P(i0+1),nullptr,256,256,1,1);
    SA.v =mkconv(ar,st,P(i0+2),nullptr,256,256,1,1);
    SA.m =mkconv(ar,st,P(i0+3),nullptr,256,256,1,1);
    SA.m1=mkconv(ar,st,P(i0+4),nullptr,512,512,1,1);
    SA.m2=mkconv(ar,st,P(i0+5),nullptr,256,512,1,1);
    SA.ln1w=P(i0+6); SA.ln1b=P(i0+7); SA.ln2w=P(i0+8); SA.ln2b=P(i0+9);
    i0=74;
    CA.q =mkconv(ar,st,P(i0+0),nullptr,256,256,1,1);
    CA.k =mkconv(ar,st,P(i0+1),nullptr,256,256,1,1);
    CA.v =mkconv(ar,st,P(i0+2),nullptr,256,256,1,1);
    CA.m =mkconv(ar,st,P(i0+3),nullptr,256,256,1,1);
    CA.m1=mkconv(ar,st,P(i0+4),nullptr,512,512,1,1);
    CA.m2=mkconv(ar,st,P(i0+5),nullptr,256,512,1,1);
    CA.ln1w=P(i0+6); CA.ln1b=P(i0+7); CA.ln2w=P(i0+8); CA.ln2b=P(i0+9);
  }
  ConvW c_off8 =mkconv(ar,st,P(84),P(85),18,256,3,3);
  ConvW c_off16=mkconv(ar,st,P(86),P(87),18,256,3,3);

  // ---- persistent tensors ----
  const size_t HW4=96*128, HW8=48*64, HW16=24*32;
  float* f      =ar.f(4ul*256*HW4);
  float* net    =ar.f(2ul*128*HW4);
  float* inp    =ar.f(2ul*128*HW4);
  float* net8   =ar.f(2ul*128*HW8);
  float* inp8   =ar.f(2ul*128*HW8);
  float* net16  =ar.f(2ul*128*HW16);
  float* inp16  =ar.f(2ul*128*HW16);
  float* f1_8   =ar.f(2ul*256*HW8);
  float* f2_8   =ar.f(2ul*256*HW8);
  float* p16a   =ar.f(2ul*256*HW16);
  float* p16b   =ar.f(2ul*256*HW16);
  float* off8   =ar.f(2ul*18*HW8);
  float* off16  =ar.f(2ul*18*HW16);
  float* f1c    =ar.f(2ul*256*HW16);
  float* f2c    =ar.f(2ul*256*HW16);
  float* flow16 =ar.f(2ul*2*HW16);
  float* flow96 =ar.f(2ul*2*HW4);
  float* flow8  =ar.f(2ul*2*HW8);
  float* flow192=ar.f(2ul*2*192*256);
  float* flow4  =ar.f(2ul*2*HW4);
  const int L16=(int)HW16, rows16=2*(int)HW16;
  float* t1 =ar.f((size_t)rows16*256);
  float* t2 =ar.f((size_t)rows16*256);
  float* t1a=ar.f((size_t)rows16*256);
  float* t2a=ar.f((size_t)rows16*256);
  TokS S;
  S.q  =ar.f((size_t)rows16*256); S.k =ar.f((size_t)rows16*256);
  S.v  =ar.f((size_t)rows16*256); S.qe=ar.f((size_t)rows16*256);
  S.ke =ar.f((size_t)rows16*256); S.ks=ar.f(2ul*256);
  S.kv =ar.f(2ul*8192);           S.ao=ar.f((size_t)rows16*256);
  S.m  =ar.f((size_t)rows16*256); S.cat=ar.f((size_t)rows16*512);
  S.h1 =ar.f((size_t)rows16*512); S.m2b=ar.f((size_t)rows16*256);
  size_t mark=ar.off;

  // ========== encoder (temp region) ==========
  {
    size_t per=2ul*3*384*512;
    float* im=ar.f(2*per);
    const size_t BIG=4ul*64*192*256;
    float* eA=ar.f(BIG); float* eB=ar.f(BIG); float* eC=ar.f(BIG); float* eD=ar.f(BIG);
    k_imnorm<<<g1(2*per),256,0,st>>>(img1,img2,im,per);
    conv(st,im,c_conv1,eA,4,384,512,2,3,3,0);
    inst(st,eA,4*64,192*256,1);
    int H=192,W=256;
    resblock(st,eA,eB,eC,eD,l1b1c1,l1b1c2,nullptr,4,H,W,1);
    resblock(st,eA,eB,eC,eD,l1b2c1,l1b2c2,nullptr,4,H,W,1);
    resblock(st,eA,eB,eC,eD,l2b1c1,l2b1c2,&l2b1d,4,H,W,2);   // -> 96x128
    resblock(st,eA,eB,eC,eD,l2b2c1,l2b2c2,nullptr,4,H,W,1);
    resblock(st,eA,eB,eC,eD,l3b1c1,l3b1c2,&l3b1d,4,H,W,1);
    resblock(st,eA,eB,eC,eD,l3b2c1,l3b2c2,nullptr,4,H,W,1);
    conv(st,eA,c_conv2,f,4,96,128,1,0,0,0);
  }
  ar.off=mark;

  const float* fmap1=f;
  const float* fmap2=f+2ul*256*HW4;

  // context features and pyramids
  k_slice_act<<<g1(2ul*128*HW4),256,0,st>>>(fmap1,net,2,256,0  ,128,(int)HW4,3); // tanh
  k_slice_act<<<g1(2ul*128*HW4),256,0,st>>>(fmap1,inp,2,256,128,128,(int)HW4,1); // relu
  k_avgpool<<<g1(2ul*128*HW8 ),256,0,st>>>(net ,net8 ,2*128,96,128,2);
  k_avgpool<<<g1(2ul*128*HW8 ),256,0,st>>>(inp ,inp8 ,2*128,96,128,2);
  k_avgpool<<<g1(2ul*128*HW16),256,0,st>>>(net ,net16,2*128,96,128,4);
  k_avgpool<<<g1(2ul*128*HW16),256,0,st>>>(inp ,inp16,2*128,96,128,4);
  k_avgpool<<<g1(2ul*256*HW8 ),256,0,st>>>(fmap1,f1_8,2*256,96,128,2);
  k_avgpool<<<g1(2ul*256*HW8 ),256,0,st>>>(fmap2,f2_8,2*256,96,128,2);
  k_avgpool<<<g1(2ul*256*HW16),256,0,st>>>(fmap1,p16a,2*256,96,128,4);
  k_avgpool<<<g1(2ul*256*HW16),256,0,st>>>(fmap2,p16b,2*256,96,128,4);
  conv(st,f1_8,c_off8 ,off8 ,2,48,64,1,1,1,2,2.f,-1.f);  // (sigmoid-.5)*2
  conv(st,p16a,c_off16,off16,2,24,32,1,1,1,2,2.f,-1.f);

  // ---- self attention + (hoisted, loop-invariant) cross attention ----
  k_posenc_tok<<<g1((size_t)rows16*256),256,0,st>>>(p16a,t1,2,256,24,32);
  k_posenc_tok<<<g1((size_t)rows16*256),256,0,st>>>(p16b,t2,2,256,24,32);
  loftr(st,SA,S,t1 ,t1 ,t1a,2,L16);
  loftr(st,SA,S,t2 ,t2 ,t2a,2,L16);
  loftr(st,CA,S,t1a,t2a,t1 ,2,L16);
  loftr(st,CA,S,t2a,t1a,t2 ,2,L16);
  k_tok2nchw<<<g1(2ul*256*HW16),256,0,st>>>(t1,f1c,2,256,24,32);
  k_tok2nchw<<<g1(2ul*256*HW16),256,0,st>>>(t2,f2c,2,256,24,32);

  // ---- update-loop temps (sized for the largest level, reused) ----
  Tmp T;
  T.corr =ar.f(2ul*36 *HW4);
  T.t256a=ar.f(2ul*256*HW4);
  T.t256b=ar.f(2ul*256*HW4);
  T.t192 =ar.f(2ul*192*HW4);
  T.t128a=ar.f(2ul*128*HW4);
  T.t128b=ar.f(2ul*128*HW4);
  T.mf   =ar.f(2ul*128*HW4);
  T.hx   =ar.f(2ul*384*HW4);
  T.z    =ar.f(2ul*128*HW4);
  T.r    =ar.f(2ul*128*HW4);
  T.q    =ar.f(2ul*128*HW4);
  T.mask =ar.f(2ul*144*HW4);
  T.dflow=ar.f(2ul*2  *HW4);

  // ========== level 1/16 ==========
  k_zero<<<g1(2ul*2*HW16),256,0,st>>>(flow16,2ul*2*HW16);
  for(int it=0; it<5; it++){
    int small=(it%2==1);
    k_corr_att<<<g1(2ul*36*HW16),256,0,st>>>(f1c,f2c,flow16,off16,T.corr,2,256,24,32,small);
    update_block(st,U,T,net16,inp16,T.corr,flow16,2,24,32,it==4);
    k_addto<<<g1(2ul*2*HW16),256,0,st>>>(flow16,T.dflow,2ul*2*HW16);
  }
  k_upsample<<<g1(2ul*2*HW4),256,0,st>>>(flow16,T.mask,flow96,2,24,32,1.f);
  k_interp<<<g1(2ul*2*HW8),256,0,st>>>(flow96,flow8,4,96,128,48,64,-0.5f);

  // ========== level 1/8 ==========
  for(int it=0; it<5; it++){
    int small=(it%2==1);
    k_corr_att<<<g1(2ul*36*HW8),256,0,st>>>(f1_8,f2_8,flow8,off8,T.corr,2,256,48,64,small);
    update_block(st,U,T,net8,inp8,T.corr,flow8,2,48,64,it==4);
    k_addto<<<g1(2ul*2*HW8),256,0,st>>>(flow8,T.dflow,2ul*2*HW8);
  }
  k_upsample<<<g1(2ul*2*192*256),256,0,st>>>(flow8,T.mask,flow192,2,48,64,1.f);
  k_interp<<<g1(2ul*2*HW4),256,0,st>>>(flow192,flow4,4,192,256,96,128,-0.5f);

  // ========== level 1/4 ==========
  for(int it=0; it<10; it++){
    int small=(it%2==1);
    k_bilin<<<g1(2ul*256*HW4),256,0,st>>>(fmap2,flow4,T.t256a,2,256,96,128);
    k_corr_patch<<<g1(2ul*36*HW4),256,0,st>>>(fmap1,T.t256a,T.corr,2,256,96,128,small);
    update_block(st,U,T,net,inp,T.corr,flow4,2,96,128,it==9);
    k_addto<<<g1(2ul*2*HW4),256,0,st>>>(flow4,T.dflow,2ul*2*HW4);
  }
  k_upsample<<<g1(2ul*2*384*512),256,0,st>>>(flow4,T.mask,(float*)d_out,2,96,128,-1.f);
#undef P
}
